// GPSLayer_27212912787996
// MI455X (gfx1250) — matrix operand order repaired and checked by reading, not compiled, not run on hardware
//
#include <hip/hip_runtime.h>
#include <hip/hip_bf16.h>
#include <math.h>

typedef _Float16 f16;
typedef __attribute__((ext_vector_type(16))) _Float16 v16h;
typedef __attribute__((ext_vector_type(8)))  float    v8f;
typedef __attribute__((ext_vector_type(4)))  _Float16 h4;

union FragU { uint4 u[2]; v16h v; _Float16 h[16]; };

#define NN 32768
#define DD 128
#define EE 524288
#define BB 64
#define SS 512
#define HHEADS 4
#define D3 384
#define D2 256

__device__ __forceinline__ v8f wmma16(v16h a, v16h b, v8f c) {
  // D = A(16x32 f16) * B(32x16 f16) + C(16x16 f32)
  return __builtin_amdgcn_wmma_f32_16x16x32_f16(false, a, false, b, (short)0, c, false, false);
}

__device__ __forceinline__ float redmax16(float v) {
  v = fmaxf(v, __shfl_xor(v, 1));
  v = fmaxf(v, __shfl_xor(v, 2));
  v = fmaxf(v, __shfl_xor(v, 4));
  v = fmaxf(v, __shfl_xor(v, 8));
  return v;
}
__device__ __forceinline__ float redsum16(float v) {
  v += __shfl_xor(v, 1);
  v += __shfl_xor(v, 2);
  v += __shfl_xor(v, 4);
  v += __shfl_xor(v, 8);
  return v;
}

// ---------------- fp32 -> fp16 convert (vectorized by 4) ----------------
__global__ void cvt_f32_f16(const float* __restrict__ src, f16* __restrict__ dst, int n4) {
  int i = blockIdx.x * blockDim.x + threadIdx.x;
  if (i >= n4) return;
  float4 v = ((const float4*)src)[i];
  h4 o; o.x = (f16)v.x; o.y = (f16)v.y; o.z = (f16)v.z; o.w = (f16)v.w;
  ((h4*)dst)[i] = o;
}

// ---------------- generic f16 WMMA GEMM: out = A[M,K] @ W[N,K]^T + bias ----------------
#define GM_F32        0
#define GM_RELU_F32F16 1
#define GM_F16        2
#define GM_RELU_F16   3
#define GM_BN_RES     4

__global__ __launch_bounds__(256) void gemm_f16(
    const f16* __restrict__ A, const f16* __restrict__ W,
    const float* __restrict__ bias, int Nout, int K, int mode,
    float* __restrict__ out32, f16* __restrict__ out16,
    const float* __restrict__ res, const float* __restrict__ bng,
    const float* __restrict__ bnb)
{
  const int lane = threadIdx.x & 31;
  const int wave = threadIdx.x >> 5;
  const int tiles_n = Nout >> 4;
  const int tile = blockIdx.x * 8 + wave;
  const int tm = tile / tiles_n;
  const int tn = tile - tm * tiles_n;
  const int g  = lane >> 4;      // lane group (0: lanes 0-15, 1: lanes 16-31)
  const int ln = lane & 15;

  const int arow = tm * 16 + ln;   // A-fragment row (M = lane%16 for both groups)
  const int ncol = tn * 16 + ln;   // B-fragment / C column

  FragU af, bf;
  v8f acc = {};
  for (int k0 = 0; k0 < K; k0 += 32) {
    // A 16x32 f16 fragment: lane holds K = k0+8g..+7 and k0+16+8g..+7 (ISA split runs)
    const f16* ap = A + (size_t)arow * K + k0 + 8 * g;
    af.u[0] = *(const uint4*)ap;
    af.u[1] = *(const uint4*)(ap + 16);
    // B 32x16 f16 fragment: lane = column n, holds K = k0+8g..+7 and k0+16+8g..+7 (ISA split runs)
    const f16* wp = W + (size_t)ncol * K + k0 + 8 * g;
    bf.u[0] = *(const uint4*)wp;
    bf.u[1] = *(const uint4*)(wp + 16);
    acc = wmma16(af.v, bf.v, acc);
  }

  const float bval = bias[ncol];
  const float bns  = rsqrtf(1.0f + 1e-5f);
  #pragma unroll
  for (int r = 0; r < 8; ++r) {
    const int row = tm * 16 + r + 8 * g;           // C layout: M = r + 8*group
    const size_t idx = (size_t)row * Nout + ncol;
    float v = acc[r] + bval;
    if (mode == GM_RELU_F32F16) {
      v = fmaxf(v, 0.f); out32[idx] = v; out16[idx] = (f16)v;
    } else if (mode == GM_F16) {
      out16[idx] = (f16)v;
    } else if (mode == GM_RELU_F16) {
      out16[idx] = (f16)fmaxf(v, 0.f);
    } else if (mode == GM_BN_RES) {
      float t = res[idx] + v;
      out32[idx] = t * (bng[ncol] * bns) + bnb[ncol];
    } else {
      out32[idx] = v;
    }
  }
}

// ---------------- edge gather/gate/scatter: agg[dst] += sigmoid(k[dst]+q[src])*v[src] ----------------
__global__ __launch_bounds__(256) void edge_kernel(
    const long long* __restrict__ ei,
    const f16* __restrict__ kh, const f16* __restrict__ qh,
    const f16* __restrict__ vh, float* __restrict__ agg)
{
  const int wave = threadIdx.x >> 5;
  const int lane = threadIdx.x & 31;
  const long long e = (long long)blockIdx.x * 8 + wave;
  const long long src = ei[e];
  const long long dst = ei[(long long)EE + e];
  const h4 kv = *(const h4*)(kh + dst * DD + lane * 4);
  const h4 qv = *(const h4*)(qh + src * DD + lane * 4);
  const h4 vv = *(const h4*)(vh + src * DD + lane * 4);
  float* ap = agg + dst * DD + lane * 4;
  #pragma unroll
  for (int j = 0; j < 4; ++j) {
    float gsum = (float)kv[j] + (float)qv[j];
    float sg = 1.0f / (1.0f + __expf(-gsum));
    atomicAdd(ap + j, sg * (float)vv[j]);
  }
}

// ---------------- flash attention per (graph, head, 16-row q tile) ----------------
// qkv layout per node row (384 f16): [q(128) | k(128) | v(128)], head h uses cols h*32..h*32+31
__global__ __launch_bounds__(256) void attn_kernel(const f16* __restrict__ qkv,
                                                   f16* __restrict__ ctx)
{
  __shared__ __align__(16) _Float16 Pld[8][16][32];   // per-wave P tile (16 q rows x 32 keys)
  const int lane = threadIdx.x & 31;
  const int wave = threadIdx.x >> 5;
  const int g  = lane >> 4;
  const int ln = lane & 15;

  const int qblk = blockIdx.x & 3;        // 4 blocks per (b,h): 128 q rows each
  const int bh   = blockIdx.x >> 2;
  const int b    = bh >> 2;               // HHEADS == 4
  const int hh   = bh & 3;
  const int qbase = qblk * 128 + wave * 16;

  const f16* base = qkv + (size_t)b * SS * D3;

  // Q fragment (A 16x32): row = ln, split K runs
  FragU qa;
  {
    const f16* p = base + (size_t)(qbase + ln) * D3 + hh * 32 + 8 * g;
    qa.u[0] = *(const uint4*)p;
    qa.u[1] = *(const uint4*)(p + 16);
  }

  float mrow[8], lrow[8];
  #pragma unroll
  for (int r = 0; r < 8; ++r) { mrow[r] = -1e30f; lrow[r] = 0.f; }
  v8f o0 = {}, o1 = {};
  const float scale = 0.17677669529663687f;   // 1/sqrt(32)

  for (int kt = 0; kt < 16; ++kt) {           // 32 keys per step
    const int keybase = kt * 32;
    // K fragments (B 32x16): lane = key column, split K(hd) runs
    FragU ka0, ka1;
    {
      const f16* p = base + (size_t)(keybase + ln) * D3 + DD + hh * 32 + 8 * g;
      ka0.u[0] = *(const uint4*)p;
      ka0.u[1] = *(const uint4*)(p + 16);
      p += (size_t)16 * D3;
      ka1.u[0] = *(const uint4*)p;
      ka1.u[1] = *(const uint4*)(p + 16);
    }
    v8f zz = {};
    v8f s0 = wmma16(qa.v, ka0.v, zz);
    v8f s1 = wmma16(qa.v, ka1.v, zz);

    #pragma unroll
    for (int r = 0; r < 8; ++r) {
      float a = s0[r] * scale;
      float c = s1[r] * scale;
      float t = redmax16(fmaxf(a, c));
      float nm = fmaxf(mrow[r], t);
      float alpha = __expf(mrow[r] - nm);
      float p0 = __expf(a - nm);
      float p1 = __expf(c - nm);
      float rs = redsum16(p0 + p1);
      lrow[r] = lrow[r] * alpha + rs;
      mrow[r] = nm;
      o0[r] *= alpha;
      o1[r] *= alpha;
      const int row = r + 8 * g;
      Pld[wave][row][ln]      = (_Float16)p0;
      Pld[wave][row][16 + ln] = (_Float16)p1;
    }
    __syncthreads();

    // P fragment (A 16x32) from LDS, split K runs
    FragU pa;
    pa.u[0] = *(const uint4*)&Pld[wave][ln][8 * g];
    pa.u[1] = *(const uint4*)&Pld[wave][ln][16 + 8 * g];

    // V fragments (B 32x16): lane = output dim col; halves walk key rows
    FragU va0, va1;
    #pragma unroll
    for (int t2 = 0; t2 < 16; ++t2) {
      const int k = (t2 < 8) ? (8 * g + t2) : (16 + 8 * g + (t2 - 8));
      const f16* vp = base + (size_t)(keybase + k) * D3 + 2 * DD + hh * 32;
      va0.h[t2] = vp[ln];
      va1.h[t2] = vp[16 + ln];
    }
    o0 = wmma16(pa.v, va0.v, o0);
    o1 = wmma16(pa.v, va1.v, o1);
    __syncthreads();
  }

  #pragma unroll
  for (int r = 0; r < 8; ++r) {
    const int row = r + 8 * g;
    const float inv = 1.0f / lrow[r];
    const size_t ob = (size_t)(b * SS + qbase + row) * DD + hh * 32;
    ctx[ob + ln]      = (f16)(o0[r] * inv);
    ctx[ob + 16 + ln] = (f16)(o1[r] * inv);
  }
}

// ---------------- fused BN/residual combine ----------------
__global__ void combine_kernel(const float* __restrict__ h_in1, const float* __restrict__ agg,
                               const float* __restrict__ skip, const float* __restrict__ outp,
                               const float* __restrict__ g1l, const float* __restrict__ b1l,
                               const float* __restrict__ g1a, const float* __restrict__ b1a,
                               float* __restrict__ hcomb, f16* __restrict__ hcombh)
{
  const int i = blockIdx.x * blockDim.x + threadIdx.x;
  const int col = i & (DD - 1);
  const float bns = rsqrtf(1.0f + 1e-5f);
  float hl = (h_in1[i] + agg[i] + skip[i]) * (g1l[col] * bns) + b1l[col];
  float ha = (h_in1[i] + outp[i]) * (g1a[col] * bns) + b1a[col];
  float hv = hl + ha;
  hcomb[i]  = hv;
  hcombh[i] = (f16)hv;
}

extern "C" void kernel_launch(void* const* d_in, const int* in_sizes, int n_in,
                              void* d_out, int out_size, void* d_ws, size_t ws_size,
                              hipStream_t stream)
{
  const float* x        = (const float*)d_in[0];
  const long long* ei   = (const long long*)d_in[1];
  const float* Wres = (const float*)d_in[3];  const float* bres = (const float*)d_in[4];
  const float* Wk   = (const float*)d_in[5];  const float* bk   = (const float*)d_in[6];
  const float* Wq   = (const float*)d_in[7];  const float* bq   = (const float*)d_in[8];
  const float* Wv   = (const float*)d_in[9];  const float* bv   = (const float*)d_in[10];
  const float* Wsk  = (const float*)d_in[11]; const float* bsk  = (const float*)d_in[12];
  const float* g1l  = (const float*)d_in[13]; const float* b1l  = (const float*)d_in[14];
  const float* g1a  = (const float*)d_in[15]; const float* b1a  = (const float*)d_in[16];
  const float* inW  = (const float*)d_in[17]; const float* inB  = (const float*)d_in[18];
  const float* outW = (const float*)d_in[19]; const float* outB = (const float*)d_in[20];
  const float* W1   = (const float*)d_in[21]; const float* b1   = (const float*)d_in[22];
  const float* W2   = (const float*)d_in[23]; const float* b2   = (const float*)d_in[24];
  const float* g2   = (const float*)d_in[25]; const float* b2g  = (const float*)d_in[26];

  char* ws = (char*)d_ws;
  size_t off = 0;
  auto carve = [&](size_t bytes) -> char* {
    char* p = ws + off;
    off += (bytes + 255) & ~(size_t)255;
    return p;
  };
  f16*  xh      = (f16*)carve((size_t)NN * DD * 2);
  f16*  wres_h  = (f16*)carve((size_t)DD * DD * 2);
  f16*  wk_h    = (f16*)carve((size_t)DD * DD * 2);
  f16*  wq_h    = (f16*)carve((size_t)DD * DD * 2);
  f16*  wv_h    = (f16*)carve((size_t)DD * DD * 2);
  f16*  wsk_h   = (f16*)carve((size_t)DD * DD * 2);
  f16*  inw_h   = (f16*)carve((size_t)D3 * DD * 2);
  f16*  outw_h  = (f16*)carve((size_t)DD * DD * 2);
  f16*  w1_h    = (f16*)carve((size_t)D2 * DD * 2);
  f16*  w2_h    = (f16*)carve((size_t)DD * D2 * 2);
  float* h_in1  = (float*)carve((size_t)NN * DD * 4);
  f16*  h_in1_h = (f16*)carve((size_t)NN * DD * 2);
  f16*  k_h     = (f16*)carve((size_t)NN * DD * 2);
  f16*  q_h     = (f16*)carve((size_t)NN * DD * 2);
  f16*  v_h     = (f16*)carve((size_t)NN * DD * 2);
  float* skip   = (float*)carve((size_t)NN * DD * 4);
  float* agg    = (float*)carve((size_t)NN * DD * 4);
  f16*  qkv_h   = (f16*)carve((size_t)NN * D3 * 2);
  f16*  ctx_h   = (f16*)carve((size_t)NN * DD * 2);
  float* outp   = (float*)carve((size_t)NN * DD * 4);
  float* hcomb  = (float*)carve((size_t)NN * DD * 4);
  f16*  hcomb_h = (f16*)carve((size_t)NN * DD * 2);
  f16*  ffn1_h  = (f16*)carve((size_t)NN * D2 * 2);
  (void)ws_size; (void)in_sizes; (void)n_in; (void)out_size;

  // --- stage 0: conversions + zero accumulator ---
  auto cvt = [&](const float* s, f16* d, int n) {
    int n4 = n / 4;
    cvt_f32_f16<<<(n4 + 255) / 256, 256, 0, stream>>>(s, d, n4);
  };
  cvt(x, xh, NN * DD);
  cvt(Wres, wres_h, DD * DD);
  cvt(Wk,   wk_h,   DD * DD);
  cvt(Wq,   wq_h,   DD * DD);
  cvt(Wv,   wv_h,   DD * DD);
  cvt(Wsk,  wsk_h,  DD * DD);
  cvt(inW,  inw_h,  D3 * DD);
  cvt(outW, outw_h, DD * DD);
  cvt(W1,   w1_h,   D2 * DD);
  cvt(W2,   w2_h,   DD * D2);
  hipMemsetAsync(agg, 0, (size_t)NN * DD * 4, stream);

  const int tm = NN / 16;  // 2048 M tiles
  // --- stage 1: five node GEMMs from x ---
  gemm_f16<<<tm * (DD/16) / 8, 256, 0, stream>>>(xh, wres_h, bres, DD, DD, GM_RELU_F32F16,
                                                 h_in1, h_in1_h, nullptr, nullptr, nullptr);
  gemm_f16<<<tm * (DD/16) / 8, 256, 0, stream>>>(xh, wk_h, bk, DD, DD, GM_F16,
                                                 nullptr, k_h, nullptr, nullptr, nullptr);
  gemm_f16<<<tm * (DD/16) / 8, 256, 0, stream>>>(xh, wq_h, bq, DD, DD, GM_F16,
                                                 nullptr, q_h, nullptr, nullptr, nullptr);
  gemm_f16<<<tm * (DD/16) / 8, 256, 0, stream>>>(xh, wv_h, bv, DD, DD, GM_F16,
                                                 nullptr, v_h, nullptr, nullptr, nullptr);
  gemm_f16<<<tm * (DD/16) / 8, 256, 0, stream>>>(xh, wsk_h, bsk, DD, DD, GM_F32,
                                                 skip, nullptr, nullptr, nullptr, nullptr);

  // --- stage 2: edge gather/gate/scatter (bandwidth-dominant) ---
  edge_kernel<<<EE / 8, 256, 0, stream>>>(ei, k_h, q_h, v_h, agg);

  // --- stage 3: QKV projection + attention ---
  gemm_f16<<<tm * (D3/16) / 8, 256, 0, stream>>>(h_in1_h, inw_h, inB, D3, DD, GM_F16,
                                                 nullptr, qkv_h, nullptr, nullptr, nullptr);
  attn_kernel<<<BB * HHEADS * (SS / 128), 256, 0, stream>>>(qkv_h, ctx_h);
  gemm_f16<<<tm * (DD/16) / 8, 256, 0, stream>>>(ctx_h, outw_h, outB, DD, DD, GM_F32,
                                                 outp, nullptr, nullptr, nullptr, nullptr);

  // --- stage 4: fused BN/residual combine ---
  combine_kernel<<<NN * DD / 256, 256, 0, stream>>>(h_in1, agg, skip, outp,
                                                    g1l, b1l, g1a, b1a, hcomb, hcomb_h);

  // --- stage 5: FFN ---
  gemm_f16<<<tm * (D2/16) / 8, 256, 0, stream>>>(hcomb_h, w1_h, b1, D2, DD, GM_RELU_F16,
                                                 nullptr, ffn1_h, nullptr, nullptr, nullptr);
  gemm_f16<<<tm * (DD/16) / 8, 256, 0, stream>>>(ffn1_h, w2_h, b2, DD, D2, GM_BN_RES,
                                                 (float*)d_out, nullptr, hcomb, g2, b2g);
}
